// GE2ELoss_41446434406514
// MI455X (gfx1250) — matrix operand order repaired and checked by reading, not compiled, not run on hardware
//
#include <hip/hip_runtime.h>
#include <hip/hip_bf16.h>

#define N_SPK 512
#define N_UTT 64
#define DIM   1024
#define ROWS  (N_SPK * N_UTT)
#define EPS   1e-6f

typedef __attribute__((ext_vector_type(16))) __bf16 v16bf;
typedef __attribute__((ext_vector_type(8)))  float  v8f;

union Frag32B { v16bf v; uint4 q[2]; };

__device__ __forceinline__ unsigned short f2bf(float f) {
  unsigned int u = __float_as_uint(f);
  unsigned int r = 0x7FFFu + ((u >> 16) & 1u);   // round-to-nearest-even
  return (unsigned short)((u + r) >> 16);
}

// ---------------------------------------------------------------------------
// Kernel 1: per speaker n: S[n,:] = sum_m x[n,m,:], sn2[n] = ||S||^2,
//           cB[n,:] = bf16( S / ||S_centroid-guarded|| )   (unit centroid)
// ---------------------------------------------------------------------------
__global__ void __launch_bounds__(256) k_spk(const float* __restrict__ x,
                                             float* __restrict__ S,
                                             unsigned short* __restrict__ cB,
                                             float* __restrict__ Sn2) {
  const int n = blockIdx.x, t = threadIdx.x;
  const int d0 = t * 4;
  const float4* xp = (const float4*)(x + (size_t)n * N_UTT * DIM + d0);
  float a0 = 0.f, a1 = 0.f, a2 = 0.f, a3 = 0.f;
  #pragma unroll 4
  for (int m = 0; m < N_UTT; ++m) {
    float4 v = xp[m * (DIM / 4)];
    a0 += v.x; a1 += v.y; a2 += v.z; a3 += v.w;
  }
  float4 s4; s4.x = a0; s4.y = a1; s4.z = a2; s4.w = a3;
  *(float4*)(S + (size_t)n * DIM + d0) = s4;

  __shared__ float red[256];
  red[t] = a0 * a0 + a1 * a1 + a2 * a2 + a3 * a3;
  __syncthreads();
  for (int sft = 128; sft > 0; sft >>= 1) {
    if (t < sft) red[t] += red[t + sft];
    __syncthreads();
  }
  const float sn2 = red[0];
  // c = (S/64) / sqrt(max(||S/64||^2, EPS))  ==  S * inv
  const float inv = 1.0f / (64.0f * sqrtf(fmaxf(sn2 * (1.0f / 4096.0f), EPS)));
  unsigned int lo = (unsigned int)f2bf(a0 * inv) | ((unsigned int)f2bf(a1 * inv) << 16);
  unsigned int hi = (unsigned int)f2bf(a2 * inv) | ((unsigned int)f2bf(a3 * inv) << 16);
  uint2 pk; pk.x = lo; pk.y = hi;
  *(uint2*)(cB + (size_t)n * DIM + d0) = pk;
  if (t == 0) Sn2[n] = sn2;
}

// ---------------------------------------------------------------------------
// Kernel 2: per row r=(n,m): eB[r,:] = bf16(x/||x||), and analytic cos_self:
//   cos_self = ((x.S - ||x||^2)/63) / (||x|| * sqrt(max(||S-x||^2/63^2, EPS)))
// ---------------------------------------------------------------------------
__global__ void __launch_bounds__(256) k_row(const float* __restrict__ x,
                                             const float* __restrict__ S,
                                             const float* __restrict__ Sn2,
                                             unsigned short* __restrict__ eB,
                                             float* __restrict__ cself) {
  const int row = blockIdx.x, t = threadIdx.x;
  const int n = row >> 6;
  const int d0 = t * 4;
  const float4 xv = *(const float4*)(x + (size_t)row * DIM + d0);
  const float4 sv = *(const float4*)(S + (size_t)n * DIM + d0);
  float ex2 = xv.x * xv.x + xv.y * xv.y + xv.z * xv.z + xv.w * xv.w;
  float xs  = xv.x * sv.x + xv.y * sv.y + xv.z * sv.z + xv.w * sv.w;

  __shared__ float rA[256], rB[256];
  rA[t] = ex2; rB[t] = xs;
  __syncthreads();
  for (int sft = 128; sft > 0; sft >>= 1) {
    if (t < sft) { rA[t] += rA[t + sft]; rB[t] += rB[t + sft]; }
    __syncthreads();
  }
  const float ex2t = rA[0], xst = rB[0];
  const float en  = sqrtf(fmaxf(ex2t, EPS));
  const float inv = 1.0f / en;
  unsigned int lo = (unsigned int)f2bf(xv.x * inv) | ((unsigned int)f2bf(xv.y * inv) << 16);
  unsigned int hi = (unsigned int)f2bf(xv.z * inv) | ((unsigned int)f2bf(xv.w * inv) << 16);
  uint2 pk; pk.x = lo; pk.y = hi;
  *(uint2*)(eB + (size_t)row * DIM + d0) = pk;

  if (t == 0) {
    const float sn2 = Sn2[n];
    const float d2  = sn2 - 2.0f * xst + ex2t;                 // ||S - x||^2
    const float ucn = sqrtf(fmaxf(d2 * (1.0f / 3969.0f), EPS)); // ||(S-x)/63|| guarded
    cself[row] = ((xst - ex2t) * (1.0f / 63.0f)) / (en * ucn);
  }
}

// ---------------------------------------------------------------------------
// Kernel 3: fused bf16 WMMA GEMM (32768x512, K=1024) + w*s+b + diagonal
// substitution + row logsumexp + global loss sum.
// Block = 64 rows x 512 cols. 8 waves, each wave: 4 row-tiles x 4 col-tiles.
// ---------------------------------------------------------------------------
__global__ void __launch_bounds__(256) k_main(const unsigned short* __restrict__ eB,
                                              const unsigned short* __restrict__ cB,
                                              const float* __restrict__ cself,
                                              const float* __restrict__ wp,
                                              const float* __restrict__ bp,
                                              float* __restrict__ out) {
  const int lane = threadIdx.x & 31;
  const int wid  = threadIdx.x >> 5;   // 0..7, owns cols [wid*64, wid*64+64)
  const int half = lane >> 4;
  const int l15  = lane & 15;
  const int r0   = blockIdx.x * 64;    // 64-row stripe, speaker n = blockIdx.x

  v8f acc[4][4];
  #pragma unroll
  for (int rt = 0; rt < 4; ++rt)
    #pragma unroll
    for (int ct = 0; ct < 4; ++ct) {
      v8f z = {0.f, 0.f, 0.f, 0.f, 0.f, 0.f, 0.f, 0.f};
      acc[rt][ct] = z;
    }

  // K loop: 32 iterations of K-chunk 32, 16 WMMA per iteration
  for (int kb = 0; kb < DIM; kb += 32) {
    Frag32B a[4];
    #pragma unroll
    for (int rt = 0; rt < 4; ++rt) {
      // A frag (16x32 bf16): lane<16 holds K {0..7,16..23}, lane>=16 {8..15,24..31}
      const unsigned short* ap = eB + (size_t)(r0 + rt * 16 + l15) * DIM + kb + half * 8;
      a[rt].q[0] = *(const uint4*)ap;
      a[rt].q[1] = *(const uint4*)(ap + 16);
    }
    #pragma unroll
    for (int ct = 0; ct < 4; ++ct) {
      // B frag (32x16 bf16): lane n=l15, lanes 0-15 K {0..7,16..23}, lanes 16-31 K {8..15,24..31}
      const unsigned short* bpnt = cB + (size_t)(wid * 64 + ct * 16 + l15) * DIM + kb + half * 8;
      Frag32B bf;
      bf.q[0] = *(const uint4*)bpnt;
      bf.q[1] = *(const uint4*)(bpnt + 16);
      #pragma unroll
      for (int rt = 0; rt < 4; ++rt)
        acc[rt][ct] = __builtin_amdgcn_wmma_f32_16x16x32_bf16(
            false, a[rt].v, false, bf.v, (short)0, acc[rt][ct], false, false);
    }
  }

  // ---------------- epilogue: sim = w*cos + b, diag -> cos_self ----------------
  __shared__ float lred[64][8];
  __shared__ float rmax_s[64];
  __shared__ float posv[64];

  const float wv = wp[0];
  const float bv = bp[0];
  const int n_diag = blockIdx.x;                        // uniform for the stripe
  const bool ownerLane = (wid == (n_diag >> 6)) && (l15 == (n_diag & 15));
  const int ct_d = (n_diag >> 4) & 3;

  #pragma unroll
  for (int rt = 0; rt < 4; ++rt)
    #pragma unroll
    for (int ct = 0; ct < 4; ++ct)
      #pragma unroll
      for (int j = 0; j < 8; ++j) {
        float s = wv * acc[rt][ct][j] + bv;
        if (ownerLane && ct == ct_d) {
          const int row64 = rt * 16 + half * 8 + j;
          s = wv * cself[r0 + row64] + bv;
          posv[row64] = s;            // positive (diagonal) logit
        }
        acc[rt][ct][j] = s;
      }

  // pass 1: per-row max (in-lane over col tiles, shfl over 16-lane half)
  #pragma unroll
  for (int rt = 0; rt < 4; ++rt)
    #pragma unroll
    for (int j = 0; j < 8; ++j) {
      float mx = fmaxf(fmaxf(acc[rt][0][j], acc[rt][1][j]),
                       fmaxf(acc[rt][2][j], acc[rt][3][j]));
      #pragma unroll
      for (int off = 1; off < 16; off <<= 1)
        mx = fmaxf(mx, __shfl_xor(mx, off, 32));
      if (l15 == 0) lred[rt * 16 + half * 8 + j][wid] = mx;
    }
  __syncthreads();
  if (threadIdx.x < 64) {
    float m = lred[threadIdx.x][0];
    #pragma unroll
    for (int q = 1; q < 8; ++q) m = fmaxf(m, lred[threadIdx.x][q]);
    rmax_s[threadIdx.x] = m;
  }
  __syncthreads();

  // pass 2: per-row sum of exp(s - rowmax)
  #pragma unroll
  for (int rt = 0; rt < 4; ++rt)
    #pragma unroll
    for (int j = 0; j < 8; ++j) {
      const int row64 = rt * 16 + half * 8 + j;
      const float rm = rmax_s[row64];
      float se = __expf(acc[rt][0][j] - rm) + __expf(acc[rt][1][j] - rm) +
                 __expf(acc[rt][2][j] - rm) + __expf(acc[rt][3][j] - rm);
      #pragma unroll
      for (int off = 1; off < 16; off <<= 1)
        se += __shfl_xor(se, off, 32);
      if (l15 == 0) lred[row64][wid] = se;
    }
  __syncthreads();

  if (threadIdx.x < 64) {
    float s = 0.f;
    #pragma unroll
    for (int q = 0; q < 8; ++q) s += lred[threadIdx.x][q];
    const float lse = rmax_s[threadIdx.x] + __logf(s);
    atomicAdd(out, lse - posv[threadIdx.x]);
  }
}

// ---------------------------------------------------------------------------
extern "C" void kernel_launch(void* const* d_in, const int* in_sizes, int n_in,
                              void* d_out, int out_size, void* d_ws, size_t ws_size,
                              hipStream_t stream) {
  (void)in_sizes; (void)n_in; (void)out_size; (void)ws_size;
  const float* x = (const float*)d_in[0];
  const float* w = (const float*)d_in[1];
  const float* b = (const float*)d_in[2];
  float* out = (float*)d_out;

  char* ws = (char*)d_ws;
  size_t off = 0;
  float* S = (float*)(ws + off);             off += (size_t)N_SPK * DIM * sizeof(float);
  float* Sn2 = (float*)(ws + off);           off += (size_t)N_SPK * sizeof(float);
  float* cself = (float*)(ws + off);         off += (size_t)ROWS * sizeof(float);
  off = (off + 255) & ~(size_t)255;
  unsigned short* cB = (unsigned short*)(ws + off); off += (size_t)N_SPK * DIM * 2;
  unsigned short* eB = (unsigned short*)(ws + off); off += (size_t)ROWS * DIM * 2;
  // total workspace ~67.3 MB

  hipMemsetAsync(d_out, 0, sizeof(float), stream);
  k_spk<<<N_SPK, 256, 0, stream>>>(x, S, cB, Sn2);
  k_row<<<ROWS, 256, 0, stream>>>(x, S, Sn2, eB, cself);
  k_main<<<ROWS / 64, 256, 0, stream>>>(eB, cB, cself, w, b, out);
}
